// PUnit_13477607375166
// MI455X (gfx1250) — matrix operand order repaired and checked by reading, not compiled, not run on hardware
//
#include <hip/hip_runtime.h>

typedef _Float16     v16h __attribute__((ext_vector_type(16)));
typedef float        v8f  __attribute__((ext_vector_type(8)));
typedef unsigned int v8u  __attribute__((ext_vector_type(8)));

// ---------------------------------------------------------------------------
// Workspace layout (all offsets 256B-aligned):
//   wpack  : 4 layers * 4 tiles * 2 chunks * 32 lanes * 16 halfs f16 = 32768 B
//   w6pack : 1 tile * 2 chunks * 32 * 16 f16                        =  2048 B
//   w1pack : 2 chunks * 32 * 16 f32 (B-layout order of w1)          =  4096 B
//   b1pack : same for b1                                            =  4096 B
// ---------------------------------------------------------------------------
#define WPACK_OFF  0
#define W6PACK_OFF 32768
#define W1PACK_OFF 34816
#define B1PACK_OFF 38912

// Hidden-unit permutation sigma(m) = swap bits 3<->4 of m (involutive).
// With inputs of every layer remapped by sigma at weight-pack time, the WMMA
// D layout (feature 16t+8hi+v at lane-half hi) becomes *identical* to the
// B-operand layout of the next layer -> activation hand-off is a pure
// per-lane f32->f16 pack (no cross-lane ops at all).
__host__ __device__ static inline int swap34(int k) {
  return (k & ~24) | ((k & 8) << 1) | ((k & 16) >> 1);
}

static __device__ __forceinline__ unsigned pk16(float a, float b) {
  auto r = __builtin_amdgcn_cvt_pkrtz(a, b);   // v_cvt_pk_rtz_f16_f32
  return __builtin_bit_cast(unsigned, r);
}

static __device__ __forceinline__ float relu1(float x) {
  return __builtin_amdgcn_fmed3f(x, 0.0f, __builtin_inff()); // folds to v_max
}

// tanh(x) = 1 - 2/(1 + exp(2x)); v_exp_f32 computes 2^x, so scale by 2*log2(e).
// Saturates correctly at +-1 (exp2 -> inf/0), NaN propagates. ~1e-7 abs error,
// far below the f16 activation quantization already present.
static __device__ __forceinline__ float fast_tanh(float x) {
  const float K = 2.8853900817779268f;               // 2 / ln(2)
  const float e = __builtin_amdgcn_exp2f(x * K);     // v_exp_f32
  const float r = __builtin_amdgcn_rcpf(e + 1.0f);   // v_rcp_f32
  return __builtin_fmaf(-2.0f, r, 1.0f);
}

// relu + pack two D accumulators (features 32c+0..15 / +16..31) into one
// B operand for the next layer.
static __device__ __forceinline__ v16h pack_b(const v8f& lo, const v8f& hi8) {
  v8u u;
#pragma unroll
  for (int i = 0; i < 4; ++i) {
    u[i]     = pk16(relu1(lo[2 * i]),  relu1(lo[2 * i + 1]));
    u[4 + i] = pk16(relu1(hi8[2 * i]), relu1(hi8[2 * i + 1]));
  }
  return __builtin_bit_cast(v16h, u);
}

// Pre-pack weights into the per-lane WMMA A-operand layout (16-bit A 16x32,
// ISA 7.12.2): lane L holds row m'=16t+(L&15); half h holds
// klocal = h + 16*(L>=16), a contiguous k order chosen here (B uses the same one). Input index remapped by sigma.
__global__ __launch_bounds__(256) void pack_weights_kernel(
    const float* __restrict__ w1, const float* __restrict__ b1,
    const float* __restrict__ w2, const float* __restrict__ w3,
    const float* __restrict__ w4, const float* __restrict__ w5,
    const float* __restrict__ w6,
    _Float16* __restrict__ wpack, _Float16* __restrict__ w6pack,
    float* __restrict__ w1pack, float* __restrict__ b1pack)
{
  const int idx = blockIdx.x * 256 + threadIdx.x;
  if (idx < 16384) {
    // hidden layers 2..5
    const int half  = idx & 15;
    const int lane  = (idx >> 4) & 31;
    const int c     = (idx >> 9) & 1;
    const int layer = idx >> 12;
    const int klocal = half + 16 * (lane >> 4);
    const int u_in = swap34(32 * c + klocal);            // sigma remap
    const int m = 16 * ((idx >> 10) & 3) + (lane & 15);  // output unit (identity)
    const float* w = (layer == 0) ? w2 : (layer == 1) ? w3 : (layer == 2) ? w4 : w5;
    wpack[idx] = (_Float16)w[u_in * 64 + m];
  } else if (idx < 16384 + 1024) {
    // layer 6 (64 -> 16)
    const int j = idx - 16384;
    const int half = j & 15;
    const int lane = (j >> 4) & 31;
    const int c    = (j >> 9) & 1;
    const int klocal = half + 16 * (lane >> 4);
    const int u_in = swap34(32 * c + klocal);            // sigma remap
    const int m = lane & 15;
    w6pack[j] = (_Float16)w6[u_in * 16 + m];
  } else if (idx < 16384 + 1024 + 1024) {
    // layer 1 params in B-operand order; unit at B slot k is sigma(k)
    const int j = idx - 17408;
    const int half = j & 15;
    const int lane = (j >> 4) & 31;
    const int c    = j >> 9;
    const int u = swap34(32 * c + 16 * (lane >> 4) + half);
    w1pack[j] = w1[u];
    b1pack[j] = b1[u];
  }
}

// One hidden 64->64 layer for TWO pixel tiles: D(64x16) = W^T(64x64)*B(64x16)
// twice, sharing the A operands. Bias folded into accumulator init.
static __device__ __forceinline__ void hidden_layer(
    const _Float16* __restrict__ wl, const float* __restrict__ bias,
    int lane, int hi,
    v16h& q0b0, v16h& q0b1, v16h& q1b0, v16h& q1b1)
{
  v8f acc0[4], acc1[4];
#pragma unroll
  for (int t = 0; t < 4; ++t) {
    const float4 blo = *(const float4*)(bias + 16 * t + 8 * hi);
    const float4 bh2 = *(const float4*)(bias + 16 * t + 8 * hi + 4);
    v8f ci;
    ci[0] = blo.x; ci[1] = blo.y; ci[2] = blo.z; ci[3] = blo.w;
    ci[4] = bh2.x; ci[5] = bh2.y; ci[6] = bh2.z; ci[7] = bh2.w;
    const v16h a0 = *(const v16h*)(wl + ((t * 2 + 0) * 32 + lane) * 16);
    const v16h a1 = *(const v16h*)(wl + ((t * 2 + 1) * 32 + lane) * 16);
    v8f c0 = __builtin_amdgcn_wmma_f32_16x16x32_f16(false, a0, false, q0b0, (short)0, ci, false, false);
    c0     = __builtin_amdgcn_wmma_f32_16x16x32_f16(false, a1, false, q0b1, (short)0, c0, false, false);
    v8f c1 = __builtin_amdgcn_wmma_f32_16x16x32_f16(false, a0, false, q1b0, (short)0, ci, false, false);
    c1     = __builtin_amdgcn_wmma_f32_16x16x32_f16(false, a1, false, q1b1, (short)0, c1, false, false);
    acc0[t] = c0;
    acc1[t] = c1;
  }
  // D -> next-layer B: pure per-lane relu+pack thanks to sigma remap
  q0b0 = pack_b(acc0[0], acc0[1]);
  q0b1 = pack_b(acc0[2], acc0[3]);
  q1b0 = pack_b(acc1[0], acc1[1]);
  q1b1 = pack_b(acc1[2], acc1[3]);
}

__global__ __launch_bounds__(256) void sr_mlp_kernel(
    const float* __restrict__ x,
    const float* __restrict__ b2, const float* __restrict__ b3,
    const float* __restrict__ b4, const float* __restrict__ b5,
    const float* __restrict__ b6,
    const _Float16* __restrict__ wpack, const _Float16* __restrict__ w6pack,
    const float* __restrict__ w1pack, const float* __restrict__ b1pack,
    float* __restrict__ out)
{
  constexpr int W  = 256;            // reference-fixed
  constexpr int HW = 256 * 256;

  const int lane = threadIdx.x & 31;
  const int hi   = lane >> 4;
  const int wave = threadIdx.x >> 5;
  const int base = (blockIdx.x * 8 + wave) * 32;   // 2 pixel tiles per wave
  const int col  = lane & 15;
  const int p0   = base + col;
  const int p1   = p0 + 16;

  const float xv0 = x[p0];
  const float xv1 = x[p1];

  // ---- layer 1: rank-1, computed directly in (sigma-remapped) B layout ----
  v16h q0b0, q0b1, q1b0, q1b1;
#pragma unroll
  for (int c = 0; c < 2; ++c) {
    const float* wp = w1pack + (c * 32 + lane) * 16;
    const float* bp = b1pack + (c * 32 + lane) * 16;
    v16h t0, t1;
#pragma unroll
    for (int h = 0; h < 16; ++h) {
      const float wv = wp[h], bv = bp[h];
      t0[h] = (_Float16)relu1(fmaf(wv, xv0, bv));
      t1[h] = (_Float16)relu1(fmaf(wv, xv1, bv));
    }
    if (c == 0) { q0b0 = t0; q1b0 = t1; }
    else        { q0b1 = t0; q1b1 = t1; }
  }

  // ---- hidden layers 2..5 (16 WMMAs each, two independent chains) ---------
  hidden_layer(wpack + 0 * 4096, b2, lane, hi, q0b0, q0b1, q1b0, q1b1);
  hidden_layer(wpack + 1 * 4096, b3, lane, hi, q0b0, q0b1, q1b0, q1b1);
  hidden_layer(wpack + 2 * 4096, b4, lane, hi, q0b0, q0b1, q1b0, q1b1);
  hidden_layer(wpack + 3 * 4096, b5, lane, hi, q0b0, q0b1, q1b0, q1b1);

  // ---- layer 6: 64 -> 16, no activation -----------------------------------
  const float4 c0 = *(const float4*)(b6 + 8 * hi);
  const float4 c1 = *(const float4*)(b6 + 8 * hi + 4);
  v8f ci;
  ci[0] = c0.x; ci[1] = c0.y; ci[2] = c0.z; ci[3] = c0.w;
  ci[4] = c1.x; ci[5] = c1.y; ci[6] = c1.z; ci[7] = c1.w;
  const v16h a0 = *(const v16h*)(w6pack + (0 * 32 + lane) * 16);
  const v16h a1 = *(const v16h*)(w6pack + (1 * 32 + lane) * 16);
  v8f o0 = __builtin_amdgcn_wmma_f32_16x16x32_f16(false, a0, false, q0b0, (short)0, ci, false, false);
  o0     = __builtin_amdgcn_wmma_f32_16x16x32_f16(false, a1, false, q0b1, (short)0, o0, false, false);
  v8f o1 = __builtin_amdgcn_wmma_f32_16x16x32_f16(false, a0, false, q1b0, (short)0, ci, false, false);
  o1     = __builtin_amdgcn_wmma_f32_16x16x32_f16(false, a1, false, q1b1, (short)0, o1, false, false);

  // ---- pixel shuffle + tanh + store ---------------------------------------
  // All 32 pixels of this wave share (n, h): base is a multiple of 32 and
  // W = 256, so n/h/w0 come from the uniform base (scalar pipe, shifts only).
  const int n  = base >> 16;          // / HW
  const int rm = base & (HW - 1);
  const int h  = rm >> 8;             // / W
  const int w0 = rm & (W - 1);        // multiple of 32
  float* outn = out + (size_t)n * (16u * HW);
  // lane holds features f = v + 8*hi of its pixel; f = i*4 + j.
#pragma unroll
  for (int q = 0; q < 2; ++q) {
    const v8f& oc = q ? o1 : o0;
    const int wcol = w0 + col + 16 * q;      // pixel column, same row for all
#pragma unroll
    for (int g = 0; g < 2; ++g) {
      const int i = 2 * hi + g;
      float4 o;
      o.x = fast_tanh(oc[4 * g + 0]);
      o.y = fast_tanh(oc[4 * g + 1]);
      o.z = fast_tanh(oc[4 * g + 2]);
      o.w = fast_tanh(oc[4 * g + 3]);
      *(float4*)(outn + (size_t)(h * 4 + i) * (4 * W) + 4 * wcol) = o;
    }
  }
}

extern "C" void kernel_launch(void* const* d_in, const int* in_sizes, int n_in,
                              void* d_out, int out_size, void* d_ws, size_t ws_size,
                              hipStream_t stream) {
  const float* x  = (const float*)d_in[0];
  const float* w1 = (const float*)d_in[1];
  const float* b1 = (const float*)d_in[2];
  const float* w2 = (const float*)d_in[3];
  const float* b2 = (const float*)d_in[4];
  const float* w3 = (const float*)d_in[5];
  const float* b3 = (const float*)d_in[6];
  const float* w4 = (const float*)d_in[7];
  const float* b4 = (const float*)d_in[8];
  const float* w5 = (const float*)d_in[9];
  const float* b5 = (const float*)d_in[10];
  const float* w6 = (const float*)d_in[11];
  const float* b6 = (const float*)d_in[12];

  char* ws = (char*)d_ws;
  _Float16* wpack  = (_Float16*)(ws + WPACK_OFF);
  _Float16* w6pack = (_Float16*)(ws + W6PACK_OFF);
  float*    w1pack = (float*)(ws + W1PACK_OFF);
  float*    b1pack = (float*)(ws + B1PACK_OFF);

  // 16384 + 1024 + 1024 = 18432 threads = 72 blocks of 256
  pack_weights_kernel<<<72, 256, 0, stream>>>(w1, b1, w2, w3, w4, w5, w6,
                                              wpack, w6pack, w1pack, b1pack);

  const int N = in_sizes[0];    // 8*3*256*256 = 1,572,864 pixels
  const int blocks = N / 256;   // 8 waves/block * 32 pixels/wave
  sr_mlp_kernel<<<blocks, 256, 0, stream>>>(x, b2, b3, b4, b5, b6,
                                            wpack, w6pack, w1pack, b1pack,
                                            (float*)d_out);
}
